// AOG_Building_Block_85358180040994
// MI455X (gfx1250) — compile-verified, matrix operand order checked by reading, not run on hardware
//
#include <hip/hip_runtime.h>
#include <cmath>
#include <cstdint>
#include <cstring>
#include <vector>
#include <map>
#include <set>
#include <string>
#include <functional>

// Problem constants (from reference): B=16, S=64, H=1024, NH=16, dh=64
#define B_ 16
#define S_ 64
#define H_ 1024
#define NH_ 16
#define DH_ 64

// ---------------------------------------------------------------------------
// Device-side types / helpers
// ---------------------------------------------------------------------------
typedef __attribute__((ext_vector_type(16))) __bf16 v16bf_t;
typedef __attribute__((ext_vector_type(8)))  float  v8f_t;

union FragAB { unsigned u[8]; v16bf_t v; };

__device__ __forceinline__ unsigned short f2bf_rte(float f) {
  union { float f; unsigned u; } x; x.f = f;
  unsigned r = x.u + 0x7FFFu + ((x.u >> 16) & 1u);   // round-to-nearest-even
  return (unsigned short)(r >> 16);
}

__device__ __forceinline__ v8f_t wmma_bf16(v16bf_t a, v16bf_t b, v8f_t c) {
#if defined(__gfx1250__)
  // D(f32 16x16) = A(bf16 16x32) * B(bf16 32x16) + C
  return __builtin_amdgcn_wmma_f32_16x16x32_bf16(false, a, false, b, (short)0, c,
                                                 false, false);
#else
  (void)a; (void)b;
  return c;
#endif
}

// K-base for fragment dword vj (pair of consecutive bf16 K values), per the
// CDNA5 16-bit A/B fragment layout: VGPR0..3 -> K 0..7 (+khalf*8),
// VGPR4..7 -> K 16..23 (+khalf*8).
__device__ __forceinline__ int frag_kbase(int vj, int khalf) {
  return ((vj < 4) ? (2 * vj) : (16 + 2 * (vj - 4))) + khalf * 8;
}

// ---------------------------------------------------------------------------
// Kernels
// ---------------------------------------------------------------------------

// Wt[n*1024 + k] = bf16( W[k*1024 + n] )  (N-major bf16 weights)
__global__ void k_w_transpose(const float* __restrict__ W,
                              unsigned short* __restrict__ Wt) {
  int idx = blockIdx.x * 256 + threadIdx.x;           // 0 .. 1M-1
  int n = idx >> 10, k = idx & 1023;
  Wt[idx] = f2bf_rte(W[k * H_ + n]);
}

// out_bf16(B, s, H) = x[:, st:st+s, :]
__global__ void k_slice_bf(const float* __restrict__ x,
                           unsigned short* __restrict__ o, int st, int s) {
  int idx = blockIdx.x * 256 + threadIdx.x;
  int n = B_ * s * H_;
  if (idx >= n) return;
  int c = idx & (H_ - 1);
  int r = idx >> 10;               // b*s + i
  int b = r / s, i = r - b * s;
  o[idx] = f2bf_rte(x[(b * S_ + st + i) * H_ + c]);
}

// out_bf16(B, sl+sr, H) = concat(L(B,sl,H), R(B,sr,H), axis=1)
__global__ void k_concat_bf(const float* __restrict__ L,
                            const float* __restrict__ R,
                            unsigned short* __restrict__ o, int sl, int sr) {
  int s = sl + sr;
  int idx = blockIdx.x * 256 + threadIdx.x;
  int n = B_ * s * H_;
  if (idx >= n) return;
  int c = idx & (H_ - 1);
  int r = idx >> 10;
  int b = r / s, i = r - b * s;
  float v = (i < sl) ? L[(b * sl + i) * H_ + c] : R[(b * sr + (i - sl)) * H_ + c];
  o[idx] = f2bf_rte(v);
}

// out_bf16 = (p0 + ... + p[cnt-1]) * inv   (cnt <= 7)
__global__ void k_avg_bf(const float* p0, const float* p1, const float* p2,
                         const float* p3, const float* p4, const float* p5,
                         const float* p6, unsigned short* __restrict__ o,
                         int n, int cnt, float inv) {
  int i = blockIdx.x * 256 + threadIdx.x;
  if (i >= n) return;
  float s = p0[i];
  if (cnt > 1) s += p1[i];
  if (cnt > 2) s += p2[i];
  if (cnt > 3) s += p3[i];
  if (cnt > 4) s += p4[i];
  if (cnt > 5) s += p5[i];
  if (cnt > 6) s += p6[i];
  o[i] = f2bf_rte(s * inv);
}

// C(M,1024) = scale * ( A_bf16(M,1024) @ W + bias )
// W given as N-major bf16: Wt[n*1024 + k].
// grid = (M/16, 4); block = 128 (4 waves). Each wave computes a 16x64 strip:
// 4 accumulators along N, one A-fragment reused across 4 WMMAs per K-step.
template <bool OUT_BF16>
__global__ __launch_bounds__(128) void k_gemm(
    const unsigned short* __restrict__ A, const unsigned short* __restrict__ Wt,
    const float* __restrict__ bias, void* __restrict__ Cout, float scale) {
  const int K = H_, N = H_;
  int wave = threadIdx.x >> 5, lane = threadIdx.x & 31;
  int lane16 = lane & 15, khalf = lane >> 4;
  int mtile = blockIdx.x * 16;
  int nbase = blockIdx.y * 256 + wave * 64;           // 4 tiles of 16 cols
  const unsigned short* arow = A + (size_t)(mtile + lane16) * K;
  const unsigned short* brow = Wt + (size_t)(nbase + lane16) * K;

  v8f_t acc0 = {}, acc1 = {}, acc2 = {}, acc3 = {};
  for (int kk = 0; kk < K; kk += 32) {
    FragAB a, b0, b1, b2, b3;
#pragma unroll
    for (int vj = 0; vj < 8; vj++) {
      int kb = frag_kbase(vj, khalf) + kk;
      a.u[vj]  = *(const unsigned*)(arow + kb);
      b0.u[vj] = *(const unsigned*)(brow + kb);
      b1.u[vj] = *(const unsigned*)(brow + 16 * K + kb);
      b2.u[vj] = *(const unsigned*)(brow + 32 * K + kb);
      b3.u[vj] = *(const unsigned*)(brow + 48 * K + kb);
    }
    if (kk + 32 < K) {                       // gfx1250 global_prefetch path
      __builtin_prefetch(arow + kk + 32, 0, 1);
      __builtin_prefetch(brow + kk + 32, 0, 1);
    }
    acc0 = wmma_bf16(a.v, b0.v, acc0);
    acc1 = wmma_bf16(a.v, b1.v, acc1);
    acc2 = wmma_bf16(a.v, b2.v, acc2);
    acc3 = wmma_bf16(a.v, b3.v, acc3);
  }

  v8f_t accs[4] = {acc0, acc1, acc2, acc3};
#pragma unroll
  for (int t = 0; t < 4; t++) {
    int n = nbase + t * 16 + lane16;
    float bn = bias[n];
#pragma unroll
    for (int vj = 0; vj < 8; vj++) {
      int m = mtile + vj + 8 * khalf;        // C/D layout: VGPR v -> row v (+8 hi)
      float val = (accs[t][vj] + bn) * scale;
      if (OUT_BF16)
        ((unsigned short*)Cout)[(size_t)m * N + n] = f2bf_rte(val);
      else
        ((float*)Cout)[(size_t)m * N + n] = val;
    }
  }
}

// Fused per-(batch,head) attention: Ob = bf16( softmax(Q K^T) V )
// Q/K/V are bf16 (B*s, 1024); Q already scaled by 1/sqrt(dh) (folded into the
// Q-projection GEMM store). Head h occupies columns [h*64, h*64+64).
// One block of 256 threads (8 waves) per (b,h). Everything padded to 64 in LDS.
__global__ __launch_bounds__(256) void k_attn(
    const unsigned short* __restrict__ Qb, const unsigned short* __restrict__ Kb,
    const unsigned short* __restrict__ Vb, unsigned short* __restrict__ Ob,
    int s) {
  __shared__ unsigned short Qs[64 * 64];   // [query][feat]
  __shared__ unsigned short Ks[64 * 64];   // [key][feat]
  __shared__ unsigned short Vts[64 * 64];  // [feat][key]  (V transposed)
  __shared__ float          Ss[64 * 64];   // scores / exp
  __shared__ unsigned short Ps[64 * 64];   // softmax probs bf16

  int bh = blockIdx.x;
  int b = bh >> 4, h = bh & 15;
  int tid = threadIdx.x;

  // Stage + zero-pad
  for (int idx = tid; idx < 64 * 64; idx += 256) {
    int i = idx >> 6, d = idx & 63;
    unsigned short q = 0, k = 0, v = 0;
    if (i < s) {
      size_t row = (size_t)(b * s + i) * H_ + h * DH_ + d;
      q = Qb[row];
      k = Kb[row];
      v = Vb[row];
    }
    Qs[idx] = q;
    Ks[idx] = k;
    Vts[d * 64 + i] = v;
  }
  __syncthreads();

  int wave = tid >> 5, lane = tid & 31;
  int lane16 = lane & 15, khalf = lane >> 4;

  // S = Q K^T  (64x64 output, K-dim = dh = 64 -> 2 WMMA steps per tile)
  for (int t = wave; t < 16; t += 8) {
    int ti = t >> 2, tj = t & 3;
    v8f_t acc = {};
    int m = ti * 16 + lane16;
    int n = tj * 16 + lane16;
    for (int kk = 0; kk < 64; kk += 32) {
      FragAB a, bb;
#pragma unroll
      for (int vj = 0; vj < 8; vj++) {
        int kb = frag_kbase(vj, khalf) + kk;
        a.u[vj]  = *(const unsigned*)&Qs[m * 64 + kb];
        bb.u[vj] = *(const unsigned*)&Ks[n * 64 + kb];
      }
      acc = wmma_bf16(a.v, bb.v, acc);
    }
#pragma unroll
    for (int vj = 0; vj < 8; vj++) {
      int mm = ti * 16 + vj + 8 * khalf;
      Ss[mm * 64 + n] = acc[vj];
    }
  }
  __syncthreads();

  // Row softmax over valid keys j < s; zero probs for padding
  if (tid < 64) {
    int i = tid;
    float mx = -1e30f;
    for (int j = 0; j < s; j++) mx = fmaxf(mx, Ss[i * 64 + j]);
    float sum = 0.f;
    for (int j = 0; j < s; j++) {
      float e = __expf(Ss[i * 64 + j] - mx);
      sum += e;
      Ss[i * 64 + j] = e;
    }
    float inv = 1.0f / sum;
    for (int j = 0; j < 64; j++)
      Ps[i * 64 + j] = (j < s) ? f2bf_rte(Ss[i * 64 + j] * inv)
                               : (unsigned short)0;
  }
  __syncthreads();

  // O = P V  (64 queries x 64 feats, K-dim = keys padded to 64)
  for (int t = wave; t < 16; t += 8) {
    int ti = t >> 2, tj = t & 3;
    v8f_t acc = {};
    int m = ti * 16 + lane16;
    int n = tj * 16 + lane16;   // feature column
    for (int kk = 0; kk < 64; kk += 32) {
      FragAB a, bb;
#pragma unroll
      for (int vj = 0; vj < 8; vj++) {
        int kb = frag_kbase(vj, khalf) + kk;
        a.u[vj]  = *(const unsigned*)&Ps[m * 64 + kb];
        bb.u[vj] = *(const unsigned*)&Vts[n * 64 + kb];
      }
      acc = wmma_bf16(a.v, bb.v, acc);
    }
#pragma unroll
    for (int vj = 0; vj < 8; vj++) {
      int mm = ti * 16 + vj + 8 * khalf;
      if (mm < s)
        Ob[(size_t)(b * s + mm) * H_ + h * DH_ + n] = f2bf_rte(acc[vj]);
    }
  }
}

// ---------------------------------------------------------------------------
// Host: faithful port of construct()/reconstruct() + launch sequence
// ---------------------------------------------------------------------------
namespace aog {

enum { T = 0, AND = 1, OR = 2 };

struct Node {
  int type = T;
  int start = 0, end = 0;          // t / or
  int ls = 0, le = 0, rs = 0, re = 0;  // and
  int depth = 0;
  std::string hk;
  std::vector<int> children;
};

static std::vector<Node> construct(int n) {
  std::vector<Node> st;
  std::map<std::string, int> hist;
  int max_depth = (int)std::ceil(std::log((double)n) / 2.0);
  int max_num_nodes = 2 * n;

  Node root;
  root.type = OR; root.start = 0; root.end = n - 1; root.depth = 0;
  root.hk = "or_0_" + std::to_string(n - 1);
  st.push_back(root);
  hist[root.hk] = 0;
  int sum = 1, idx = 0;

  while (idx < sum) {
    Node nd = st[idx];  // copy (push_back may reallocate)
    std::vector<int> ch;
    if (nd.type == OR) {
      std::string hk = "t_" + std::to_string(nd.start) + "_" + std::to_string(nd.end);
      auto it = hist.find(hk);
      if (it != hist.end()) {
        ch.push_back(it->second);
      } else {
        Node t; t.type = T; t.start = nd.start; t.end = nd.end;
        t.depth = nd.depth; t.hk = hk;
        st.push_back(t); hist[hk] = sum; ch.push_back(sum); sum++;
      }
      std::set<int> ms;
      ms.insert(nd.start);
      int pmax = (int)std::floor(std::log2((double)(nd.end - nd.start)));
      for (int p = 0; p < pmax; p++) ms.insert(nd.start + (1 << p));
      for (int m : ms) {
        if (m < nd.start || m >= nd.end || sum > max_num_nodes) continue;
        std::string ahk = "and_" + std::to_string(nd.start) + "_" +
                          std::to_string(m) + "_" + std::to_string(nd.end);
        auto ai = hist.find(ahk);
        if (ai != hist.end()) {
          ch.push_back(ai->second);
        } else {
          Node a; a.type = AND;
          a.ls = nd.start; a.le = m; a.rs = m + 1; a.re = nd.end;
          a.depth = nd.depth + 1; a.hk = ahk;
          st.push_back(a); hist[ahk] = sum; ch.push_back(sum); sum++;
        }
      }
      st[idx].children = ch;
    } else if (nd.type == AND) {
      bool force_t = (nd.depth == max_depth) && (max_num_nodes + 2 < sum);
      auto mk = [&](int s0, int e0, int depth, bool ft, std::string& key, Node& node) {
        if (s0 == e0 || ft) {
          key = (s0 == e0) ? ("t_" + std::to_string(s0))
                           : ("t_" + std::to_string(s0) + "_" + std::to_string(e0));
          node.type = T; node.start = s0; node.end = e0; node.depth = depth; node.hk = key;
        } else {
          key = "or_" + std::to_string(s0) + "_" + std::to_string(e0);
          node.type = OR; node.start = s0; node.end = e0; node.depth = depth + 1; node.hk = key;
        }
      };
      std::string lk, rk; Node lc, rc;
      mk(nd.ls, nd.le, nd.depth, force_t, lk, lc);
      mk(nd.rs, nd.re, nd.depth, force_t, rk, rc);
      auto li = hist.find(lk);
      if (li != hist.end()) ch.push_back(li->second);
      else { st.push_back(lc); hist[lk] = sum; ch.push_back(sum); sum++; }
      auto ri = hist.find(rk);
      if (ri != hist.end()) ch.push_back(ri->second);
      else { st.push_back(rc); hist[rk] = sum; ch.push_back(sum); sum++; }
      st[idx].children = ch;
    }
    idx++;
  }
  return st;
}

static std::vector<Node> reconstruct(const std::vector<Node>& old) {
  std::vector<Node> ns;
  std::map<std::string, int> hist;
  std::function<int(int)> dfs = [&](int oi) -> int {
    auto it = hist.find(old[oi].hk);
    if (it != hist.end()) return it->second;
    Node nd = old[oi];
    if (nd.type != T)
      for (size_t c = 0; c < nd.children.size(); c++)
        nd.children[c] = dfs(old[oi].children[c]);
    int id = (int)ns.size();
    hist[nd.hk] = id;
    ns.push_back(nd);
    return id;
  };
  dfs(0);
  return ns;
}

static inline int node_seq(const Node& nd) {
  if (nd.type == AND) return (nd.le - nd.ls + 1) + (nd.re - nd.rs + 1);
  return nd.end - nd.start + 1;
}

}  // namespace aog

extern "C" void kernel_launch(void* const* d_in, const int* in_sizes, int n_in,
                              void* d_out, int out_size, void* d_ws, size_t ws_size,
                              hipStream_t stream) {
  (void)in_sizes; (void)n_in; (void)out_size; (void)ws_size;

  const float* x  = (const float*)d_in[0];
  const float* Wq = (const float*)d_in[1];
  const float* bq = (const float*)d_in[2];
  const float* Wk = (const float*)d_in[3];
  const float* bk = (const float*)d_in[4];
  const float* Wv = (const float*)d_in[5];
  const float* bv = (const float*)d_in[6];
  const float* Wl = (const float*)d_in[7];
  const float* bl = (const float*)d_in[8];

  char* ws = (char*)d_ws;
  const size_t MB = 1u << 20;
  // Fixed scratch layout (all bf16 activations now)
  unsigned short* Wq_t = (unsigned short*)(ws + 0 * MB);   // 2MB each
  unsigned short* Wk_t = (unsigned short*)(ws + 2 * MB);
  unsigned short* Wv_t = (unsigned short*)(ws + 4 * MB);
  unsigned short* Wl_t = (unsigned short*)(ws + 6 * MB);
  unsigned short* Xb   = (unsigned short*)(ws + 8 * MB);   // 2MB
  unsigned short* Qb   = (unsigned short*)(ws + 10 * MB);  // 2MB
  unsigned short* Kb   = (unsigned short*)(ws + 12 * MB);  // 2MB
  unsigned short* Vb   = (unsigned short*)(ws + 14 * MB);  // 2MB
  unsigned short* Ob   = (unsigned short*)(ws + 16 * MB);  // 2MB
  size_t data_base = 18 * MB;

  // One-time (per call): bf16 N-major weights
  {
    int blocks = (H_ * H_) / 256;
    k_w_transpose<<<blocks, 256, 0, stream>>>(Wq, Wq_t);
    k_w_transpose<<<blocks, 256, 0, stream>>>(Wk, Wk_t);
    k_w_transpose<<<blocks, 256, 0, stream>>>(Wv, Wv_t);
    k_w_transpose<<<blocks, 256, 0, stream>>>(Wl, Wl_t);
  }

  // Build the AOG structure (deterministic host work)
  std::vector<aog::Node> structure = aog::reconstruct(aog::construct(S_));
  int nnodes = (int)structure.size();

  // Bump-allocate per-node fp32 activations; root writes directly to d_out
  std::vector<float*> data(nnodes, nullptr);
  size_t off = data_base;
  for (int i = 0; i < nnodes; i++) {
    if (i == nnodes - 1) { data[i] = (float*)d_out; continue; }
    data[i] = (float*)(ws + off);
    off += (size_t)B_ * aog::node_seq(structure[i]) * H_ * sizeof(float);
    off = (off + 255) & ~(size_t)255;
  }

  const float qscale = 0.125f;  // 1/sqrt(dh), folded into Q-projection store

  auto attn_lin = [&](int s, float* out) {
    int M = B_ * s;
    dim3 gg(M / 16, 4);
    k_gemm<true><<<gg, 128, 0, stream>>>(Xb, Wq_t, bq, Qb, qscale);
    k_gemm<true><<<gg, 128, 0, stream>>>(Xb, Wk_t, bk, Kb, 1.0f);
    k_gemm<true><<<gg, 128, 0, stream>>>(Xb, Wv_t, bv, Vb, 1.0f);
    k_attn<<<B_ * NH_, 256, 0, stream>>>(Qb, Kb, Vb, Ob, s);
    k_gemm<false><<<gg, 128, 0, stream>>>(Ob, Wl_t, bl, out, 1.0f);
  };

  for (int i = 0; i < nnodes; i++) {
    const aog::Node& nd = structure[i];
    if (nd.type == aog::T && nd.start == nd.end) {
      // lin(x[:, s:s+1])
      int n = B_ * 1 * H_;
      int blocks = (n + 255) / 256;
      k_slice_bf<<<blocks, 256, 0, stream>>>(x, Xb, nd.start, 1);
      dim3 gg(B_ / 16, 4);
      k_gemm<false><<<gg, 128, 0, stream>>>(Xb, Wl_t, bl, data[i], 1.0f);
      continue;
    }

    int s = aog::node_seq(nd);
    int n = B_ * s * H_;
    int blocks = (n + 255) / 256;

    if (nd.type == aog::T) {
      k_slice_bf<<<blocks, 256, 0, stream>>>(x, Xb, nd.start, s);
    } else if (nd.type == aog::AND) {
      int lc = nd.children[0], rc = nd.children[1];
      int sl = aog::node_seq(structure[lc]);
      int sr = aog::node_seq(structure[rc]);
      k_concat_bf<<<blocks, 256, 0, stream>>>(data[lc], data[rc], Xb, sl, sr);
    } else {  // OR: average of children then attn+lin
      int cnt = (int)nd.children.size();
      const float* p[7];
      for (int c = 0; c < 7; c++)
        p[c] = (c < cnt) ? data[nd.children[c]] : data[nd.children[0]];
      k_avg_bf<<<blocks, 256, 0, stream>>>(p[0], p[1], p[2], p[3], p[4], p[5],
                                           p[6], Xb, n, cnt, 1.0f / (float)cnt);
    }
    attn_lin(s, data[i]);
  }
}
